// TriplaneSynthesisBlock_61753039781916
// MI455X (gfx1250) — matrix operand order repaired and checked by reading, not compiled, not run on hardware
//
#include <hip/hip_runtime.h>
#include <hip/hip_bf16.h>

typedef __attribute__((ext_vector_type(16))) __bf16 v16bf;
typedef __attribute__((ext_vector_type(8)))  float  v8f;

#define RDIM 128
#define XP   130          // padded (halo) spatial dim
#define BDIM 2
#define SQRT2F 1.41421356237309515f

__device__ __forceinline__ unsigned short f2bf(float f) {
    unsigned int u = __float_as_uint(f);
    u += 0x7fffu + ((u >> 16) & 1u);   // round-to-nearest-even
    return (unsigned short)(u >> 16);
}

union HalfTile { uint4 u[2]; v16bf v; };

// ---------------------------------------------------------------------------
// styles[b][c] = (dot(wlat[b,layer,:], aw[c,:]) / sqrt(512) + ab[c]) * extra
// ---------------------------------------------------------------------------
__global__ void fc_styles_kernel(const float* __restrict__ wlat,
                                 const float* __restrict__ aw,
                                 const float* __restrict__ ab,
                                 float* __restrict__ st,
                                 int NI, int layer, float extra) {
    int i = blockIdx.x * blockDim.x + threadIdx.x;
    if (i >= BDIM * NI) return;
    int b = i / NI, c = i % NI;
    const float* wl = wlat + (size_t)(b * 3 + layer) * 512;
    const float* ar = aw + (size_t)c * 512;
    float s = 0.f;
    #pragma unroll 8
    for (int d = 0; d < 512; ++d) s += wl[d] * ar[d];
    st[i] = (s * 0.04419417382415922f + ab[c]) * extra;
}

// ---------------------------------------------------------------------------
// dm[b][o] = rsqrt( sum_{c,k} (w[o][c][k] * st[b][c])^2 + 1e-8 )
// ---------------------------------------------------------------------------
__global__ void demod_kernel(const float* __restrict__ w,
                             const float* __restrict__ st,
                             float* __restrict__ dm) {
    int i = blockIdx.x * blockDim.x + threadIdx.x;
    if (i >= BDIM * 256) return;
    int b = i / 256, o = i % 256;
    float t = 0.f;
    for (int c = 0; c < 768; ++c) {
        float s = st[b * 768 + c];
        const float* wp = w + ((size_t)o * 768 + c) * 9;
        float a = 0.f;
        #pragma unroll
        for (int k = 0; k < 9; ++k) { float x = wp[k]; a += x * x; }
        t += a * s * s;
    }
    dm[i] = rsqrtf(t + 1e-8f);
}

// ---------------------------------------------------------------------------
// Pack modulated weight into WMMA A-fragment order (bf16).
// K order: kglobal = (kh*3+kw)*CIN3 + c.  Lane L holds M = mt*16 + (L&15),
// kbase = (L<16)?0:16;  elem e -> klocal = kbase+e.
// ---------------------------------------------------------------------------
template<int COUT, int CIN3, int KSZ, bool DEMOD>
__global__ void pack_w_kernel(const float* __restrict__ w,
                              const float* __restrict__ st,
                              const float* __restrict__ dm,
                              unsigned short* __restrict__ A) {
    constexpr int MT = COUT / 16;
    constexpr int CT = CIN3 / 32;
    constexpr int KT = KSZ * KSZ * CT;
    int tid = blockIdx.x * blockDim.x + threadIdx.x;
    if (tid >= BDIM * MT * KT * 32) return;
    int lane = tid & 31; int t = tid >> 5;
    int kt = t % KT; t /= KT;
    int mt = t % MT; int b = t / MT;
    int m = mt * 16 + (lane & 15);
    int kbase = (lane < 16) ? 0 : 16;
    float dmod = DEMOD ? dm[b * COUT + m] : 1.f;

    unsigned int packed[8];
    unsigned short h[16];
    #pragma unroll
    for (int e = 0; e < 16; ++e) {
        int kl = kbase + e;
        int kg = kt * 32 + kl;
        int kid = (KSZ == 3) ? (kg / CIN3) : 0;
        int c = kg - kid * CIN3;
        float wv = w[((size_t)m * CIN3 + c) * (KSZ * KSZ) + kid]
                   * st[b * CIN3 + c] * dmod;
        h[e] = f2bf(wv);
    }
    #pragma unroll
    for (int j = 0; j < 8; ++j)
        packed[j] = (unsigned int)h[2 * j] | ((unsigned int)h[2 * j + 1] << 16);
    uint4* dst = (uint4*)(A + (((size_t)(b * MT + mt) * KT + kt) * 512) + lane * 16);
    dst[0] = make_uint4(packed[0], packed[1], packed[2], packed[3]);
    dst[1] = make_uint4(packed[4], packed[5], packed[6], packed[7]);
}

// ---------------------------------------------------------------------------
// Row/col means per plane.  in: (B,3,256,R,R) f32.
// rm[(p*B+b)*256*128 + c*128 + y], cm[...*128 + x]
// ---------------------------------------------------------------------------
__global__ void means_kernel(const float* __restrict__ in,
                             float* __restrict__ rm,
                             float* __restrict__ cm) {
    int i = blockIdx.x * blockDim.x + threadIdx.x;
    if (i >= 6 * 256 * 128) return;
    int idx = i & 127; int t = i >> 7;
    int c = t & 255; int pb = t >> 8;
    int p = pb / BDIM, b = pb % BDIM;
    const float* base = in + (((size_t)(b * 3 + p) * 256 + c) * RDIM) * RDIM;
    float rs = 0.f, cs = 0.f;
    for (int j = 0; j < RDIM; ++j) {
        rs += base[idx * RDIM + j];
        cs += base[j * RDIM + idx];
    }
    rm[i] = rs * (1.f / 128.f);
    cm[i] = cs * (1.f / 128.f);
}

// ---------------------------------------------------------------------------
// Build concatenated, HALO-PADDED NHWC bf16 input:
//   X[(pb*130 + y)*130 + x][cc], cc in [0,768); border pixels (y/x==0 or 129)
//   are written as zeros so the conv needs no bounds checks.
//   interior (yi=y-1, xi=x-1):
//     cc<256: plane value; 256..511: rowmean[(p+2)%3][c][xi]; 512..767: colmean[(p+1)%3][c][yi]
// ---------------------------------------------------------------------------
__global__ void pack_x_kernel(const float* __restrict__ in,
                              const float* __restrict__ rm,
                              const float* __restrict__ cm,
                              unsigned short* __restrict__ X) {
    int i = blockIdx.x * blockDim.x + threadIdx.x;
    if (i >= 6 * XP * XP * 192) return;
    int c4 = i % 192; int t = i / 192;
    int x = t % XP; t /= XP;
    int y = t % XP; int pb = t / XP;
    int c0 = c4 * 4;
    unsigned int* dst = (unsigned int*)(X + (((size_t)(pb * XP + y) * XP + x) * 768 + c0));
    if (x == 0 || x == XP - 1 || y == 0 || y == XP - 1) {
        dst[0] = 0u; dst[1] = 0u;
        return;
    }
    int xi = x - 1, yi = y - 1;
    int p = pb / BDIM, b = pb % BDIM;
    float v[4];
    if (c0 < 256) {
        #pragma unroll
        for (int j = 0; j < 4; ++j)
            v[j] = in[(((size_t)(b * 3 + p) * 256 + c0 + j) * RDIM + yi) * RDIM + xi];
    } else if (c0 < 512) {
        int sp = (p + 2) % 3;
        #pragma unroll
        for (int j = 0; j < 4; ++j)
            v[j] = rm[((size_t)(sp * BDIM + b) * 256 + (c0 - 256 + j)) * RDIM + xi];
    } else {
        int sp = (p + 1) % 3;
        #pragma unroll
        for (int j = 0; j < 4; ++j)
            v[j] = cm[((size_t)(sp * BDIM + b) * 256 + (c0 - 512 + j)) * RDIM + yi];
    }
    dst[0] = (unsigned int)f2bf(v[0]) | ((unsigned int)f2bf(v[1]) << 16);
    dst[1] = (unsigned int)f2bf(v[2]) | ((unsigned int)f2bf(v[3]) << 16);
}

// Halo-layout NHWC bf16 pack of the 256-channel feature for toRGB (border zeroed).
__global__ void pack_x_rgb_kernel(const float* __restrict__ in,
                                  unsigned short* __restrict__ X) {
    int i = blockIdx.x * blockDim.x + threadIdx.x;
    if (i >= 6 * XP * XP * 64) return;
    int c4 = i % 64; int t = i / 64;
    int x = t % XP; t /= XP;
    int y = t % XP; int pb = t / XP;
    int c0 = c4 * 4;
    unsigned int* dst = (unsigned int*)(X + (((size_t)(pb * XP + y) * XP + x) * 256 + c0));
    if (x == 0 || x == XP - 1 || y == 0 || y == XP - 1) {
        dst[0] = 0u; dst[1] = 0u;
        return;
    }
    int xi = x - 1, yi = y - 1;
    int p = pb / BDIM, b = pb % BDIM;
    float v[4];
    #pragma unroll
    for (int j = 0; j < 4; ++j)
        v[j] = in[(((size_t)(b * 3 + p) * 256 + c0 + j) * RDIM + yi) * RDIM + xi];
    dst[0] = (unsigned int)f2bf(v[0]) | ((unsigned int)f2bf(v[1]) << 16);
    dst[1] = (unsigned int)f2bf(v[2]) | ((unsigned int)f2bf(v[3]) << 16);
}

// ---------------------------------------------------------------------------
// Implicit-GEMM conv via v_wmma_f32_16x16x32_bf16 on halo-padded NHWC input.
// Wave: 2 M-tiles (32 couts) x 8 N-subtiles (128 pixels) => per K-step:
//   2 A loads + 8 B loads feed 16 WMMAs (each B fragment reused 2x).
// Grid: (MT/2/wavesPerBlock, R, 6).  TRI: +noise, +bias, lrelu*sqrt2, clamp.
// ---------------------------------------------------------------------------
template<int COUT, int CIN3, int KSZ, bool TRI>
__global__ __launch_bounds__(128)
void conv_wmma_kernel(const unsigned short* __restrict__ A,
                      const unsigned short* __restrict__ X,
                      const float* __restrict__ nc,
                      const float* __restrict__ ns,
                      const float* __restrict__ bias,
                      float* __restrict__ out) {
    constexpr int MT = COUT / 16;
    constexpr int CT = CIN3 / 32;
    constexpr int KT = KSZ * KSZ * CT;

    const int lane = threadIdx.x & 31;
    const int wv   = threadIdx.x >> 5;
    const int n    = lane & 15;
    const int hi   = lane >> 4;
    const int mg   = blockIdx.x * (blockDim.x >> 5) + wv;   // m-group (2 tiles)
    const int mt0  = mg * 2;
    const int pb   = blockIdx.z;
    const int y    = blockIdx.y;
    const int b    = pb % BDIM;
    const int p    = pb / BDIM;

    v8f acc0[8] = {};
    v8f acc1[8] = {};

    const uint4* __restrict__ Ap0 =
        (const uint4*)(A + ((size_t)(b * MT + mt0) * KT) * 512) + lane * 2;
    const uint4* __restrict__ Ap1 = Ap0 + (size_t)KT * 64;

    for (int kid = 0; kid < KSZ * KSZ; ++kid) {
        const int dy = (KSZ == 3) ? (kid / 3 - 1) : 0;
        const int dx = (KSZ == 3) ? (kid % 3 - 1) : 0;
        // halo layout: interior pixel (y,x) lives at (y+1, x+1)
        const unsigned short* __restrict__ Xbase =
            X + (((size_t)(pb * XP + (y + 1 + dy)) * XP) + (1 + dx + n)) * CIN3 + hi * 16;

        for (int ct = 0; ct < CT; ++ct) {
            HalfTile a0, a1;
            a0.u[0] = Ap0[0];
            a0.u[1] = Ap0[1];
            a1.u[0] = Ap1[0];
            a1.u[1] = Ap1[1];
            __builtin_prefetch((const void*)(Ap0 + 64), 0, 1);  // next weight tiles
            __builtin_prefetch((const void*)(Ap1 + 64), 0, 1);
            Ap0 += 64;
            Ap1 += 64;

            const unsigned short* __restrict__ Xc = Xbase + ct * 32;
            #pragma unroll
            for (int s = 0; s < 8; ++s) {
                HalfTile bf;
                const uint4* Bp = (const uint4*)(Xc + (size_t)(s * 16) * CIN3);
                bf.u[0] = Bp[0];
                bf.u[1] = Bp[1];
                acc0[s] = __builtin_amdgcn_wmma_f32_16x16x32_bf16(
                    false, a0.v, false, bf.v, (short)0, acc0[s], false, false);
                acc1[s] = __builtin_amdgcn_wmma_f32_16x16x32_bf16(
                    false, a1.v, false, bf.v, (short)0, acc1[s], false, false);
            }
        }
    }

    float nsv = TRI ? ns[0] : 0.f;
    #pragma unroll
    for (int s = 0; s < 8; ++s) {
        int px = s * 16 + n;
        float noise = TRI ? nc[y * RDIM + px] * nsv : 0.f;
        #pragma unroll
        for (int mi = 0; mi < 2; ++mi) {
            #pragma unroll
            for (int r = 0; r < 8; ++r) {
                int m = (mt0 + mi) * 16 + hi * 8 + r;
                float v = (mi == 0) ? acc0[s][r] : acc1[s][r];
                if (TRI) v += noise;
                v += bias[m];
                if (TRI) v = (v >= 0.f ? v : 0.2f * v) * SQRT2F;
                v = fminf(fmaxf(v, -256.f), 256.f);
                out[(((size_t)(b * 3 + p) * COUT + m) * RDIM + y) * RDIM + px] = v;
            }
        }
    }
}

// ---------------------------------------------------------------------------
extern "C" void kernel_launch(void* const* d_in, const int* in_sizes, int n_in,
                              void* d_out, int out_size, void* d_ws, size_t ws_size,
                              hipStream_t stream) {
    const float* x      = (const float*)d_in[0];
    const float* wlat   = (const float*)d_in[1];
    const float* c0_aw  = (const float*)d_in[2];
    const float* c0_ab  = (const float*)d_in[3];
    const float* c0_w   = (const float*)d_in[4];
    const float* c0_b   = (const float*)d_in[5];
    const float* c0_ns  = (const float*)d_in[6];
    const float* nc0    = (const float*)d_in[7];
    const float* c1_aw  = (const float*)d_in[8];
    const float* c1_ab  = (const float*)d_in[9];
    const float* c1_w   = (const float*)d_in[10];
    const float* c1_b   = (const float*)d_in[11];
    const float* c1_ns  = (const float*)d_in[12];
    const float* nc1    = (const float*)d_in[13];
    const float* rgb_aw = (const float*)d_in[14];
    const float* rgb_ab = (const float*)d_in[15];
    const float* rgb_w  = (const float*)d_in[16];
    const float* rgb_b  = (const float*)d_in[17];

    char* ws = (char*)d_ws;
    size_t off = 0;
    auto alloc = [&](size_t bytes) {
        void* pp = ws + off;
        off += (bytes + 255) & ~(size_t)255;
        return pp;
    };
    unsigned short* XX    = (unsigned short*)alloc((size_t)6 * XP * XP * 768 * 2);
    float*          feat1 = (float*)alloc((size_t)2 * 3 * 256 * 128 * 128 * 4);
    unsigned short* Aw    = (unsigned short*)alloc((size_t)2 * 256 * 6912 * 2);
    float*          st    = (float*)alloc((size_t)2 * 768 * 4);
    float*          dm    = (float*)alloc((size_t)2 * 256 * 4);
    float*          rm    = (float*)alloc((size_t)6 * 256 * 128 * 4);
    float*          cm    = (float*)alloc((size_t)6 * 256 * 128 * 4);
    unsigned short* Argb  = (unsigned short*)alloc((size_t)2 * 2 * 8 * 512 * 2);
    float*          strgb = (float*)alloc((size_t)2 * 256 * 4);

    float* out_feat = (float*)d_out;
    float* out_img  = (float*)d_out + (size_t)2 * 3 * 256 * 128 * 128;

    dim3 convGrid(2, 128, 6), convBlk(128);   // 4 waves x 2 m-tiles x 2 blocks = MT=16
    dim3 rgbGrid(1, 128, 6), rgbBlk(32);      // 1 wave covers MT=2
    const int packWTot  = 2 * 16 * (9 * 24) * 32;       // 221184
    const int packXTot  = 6 * XP * XP * 192;
    const int packXrTot = 6 * XP * XP * 64;
    const int meansTot  = 6 * 256 * 128;

    // ---- layer 0: x -> feat1 ----
    fc_styles_kernel<<<(2 * 768 + 255) / 256, 256, 0, stream>>>(wlat, c0_aw, c0_ab, st, 768, 0, 1.f);
    demod_kernel<<<2, 256, 0, stream>>>(c0_w, st, dm);
    pack_w_kernel<256, 768, 3, true><<<(packWTot + 255) / 256, 256, 0, stream>>>(c0_w, st, dm, Aw);
    means_kernel<<<(meansTot + 255) / 256, 256, 0, stream>>>(x, rm, cm);
    pack_x_kernel<<<(packXTot + 255) / 256, 256, 0, stream>>>(x, rm, cm, XX);
    conv_wmma_kernel<256, 768, 3, true><<<convGrid, convBlk, 0, stream>>>(Aw, XX, nc0, c0_ns, c0_b, feat1);

    // ---- layer 1: feat1 -> out_feat ----
    fc_styles_kernel<<<(2 * 768 + 255) / 256, 256, 0, stream>>>(wlat, c1_aw, c1_ab, st, 768, 1, 1.f);
    demod_kernel<<<2, 256, 0, stream>>>(c1_w, st, dm);
    pack_w_kernel<256, 768, 3, true><<<(packWTot + 255) / 256, 256, 0, stream>>>(c1_w, st, dm, Aw);
    means_kernel<<<(meansTot + 255) / 256, 256, 0, stream>>>(feat1, rm, cm);
    pack_x_kernel<<<(packXTot + 255) / 256, 256, 0, stream>>>(feat1, rm, cm, XX);
    conv_wmma_kernel<256, 768, 3, true><<<convGrid, convBlk, 0, stream>>>(Aw, XX, nc1, c1_ns, c1_b, out_feat);

    // ---- toRGB: out_feat -> out_img ----
    fc_styles_kernel<<<(2 * 256 + 255) / 256, 256, 0, stream>>>(wlat, rgb_aw, rgb_ab, strgb, 256, 2, 0.0625f);
    pack_w_kernel<32, 256, 1, false><<<(2 * 2 * 8 * 32 + 255) / 256, 256, 0, stream>>>(rgb_w, strgb, dm, Argb);
    pack_x_rgb_kernel<<<(packXrTot + 255) / 256, 256, 0, stream>>>(out_feat, XX);
    conv_wmma_kernel<32, 256, 1, false><<<rgbGrid, rgbBlk, 0, stream>>>(Argb, XX, nullptr, nullptr, rgb_b, out_img);
}
